// Attention_41729902248209
// MI455X (gfx1250) — matrix operand order repaired and checked by reading, not compiled, not run on hardware
//
#include <hip/hip_runtime.h>
#include <hip/hip_bf16.h>

// ---------------------------------------------------------------------------
// MI455X (gfx1250) attention block:
//   k0a: x (b,cin,n) -> Xt (b,n,cin) bf16 via LDS-tiled transpose
//   k0b: W_fc/W_last f32 -> bf16 (one-time; kills recurring cvt+2x bytes)
//   k1 : QKV GEMM (WMMA bf16) -> Q*0.125, K, V^T
//   k2 : flash attention (WMMA bf16, register-only S^T -> P^T hand-off)
//   k3 : out GEMM + bias + residual, transposed back to (b,cin,h,w)
// ---------------------------------------------------------------------------

typedef __attribute__((ext_vector_type(16))) __bf16 v16bf;
typedef __attribute__((ext_vector_type(8)))  float  v8f;

#define BS    8
#define CIN   512
#define NTOK  1024
#define HEADS 8
#define DH    64
#define HD    512      // HEADS*DH
#define QKVR  1536     // 3*HD

__device__ __forceinline__ v8f wmma_bf16(v16bf a, v16bf b, v8f c) {
  // 8 args: (neg_a, A, neg_b, B, c_mod, C, reuse_a, reuse_b)
  return __builtin_amdgcn_wmma_f32_16x16x32_bf16(false, a, false, b, (short)0, c,
                                                 false, false);
}

// A fragment (16x32 16-bit): lanes 0-15 row M=lane, K = 0..15;
// lanes 16-31 same row, K = 16..31. Two contiguous 16B loads.
__device__ __forceinline__ v16bf load_a_bf16(const __bf16* base, int ld, int row,
                                             int kbase, int lane) {
  const __bf16* p = base + (long)row * ld + kbase + ((lane >> 4) << 4);
  union { uint4 u[2]; v16bf v; } fr;
  fr.u[0] = *(const uint4*)(p);
  fr.u[1] = *(const uint4*)(p + 8);
  return fr.v;
}

// B fragment (32x16 16-bit): lane holds column N=lane%16;
// lanes 0-15: K = 0..15, lanes 16-31: K = 16..31 (one contiguous run of 16).
__device__ __forceinline__ v16bf load_b_bf16(const __bf16* base, int ld, int col,
                                             int kbase, int lane) {
  const __bf16* p = base + (long)col * ld + kbase + ((lane >> 4) << 4);
  union { uint4 u[2]; v16bf v; } fr;
  fr.u[0] = *(const uint4*)(p);
  fr.u[1] = *(const uint4*)(p + 8);
  return fr.v;
}

// ---------------------------------------------------------------------------
// Kernel 0a: LDS-tiled transpose+convert: Xt[b][i][c] = (bf16)x[b][c][i].
// 32x32 tiles, both global sides coalesced. Grid (32,16,8), block 256.
// ---------------------------------------------------------------------------
__global__ void __launch_bounds__(256)
xt_kernel(const float* __restrict__ x, __bf16* __restrict__ Xt) {
  __shared__ float tile[32][33];            // +1 pad: no bank conflicts
  const int b  = blockIdx.z;
  const int i0 = blockIdx.x << 5;
  const int c0 = blockIdx.y << 5;
  const int tx = threadIdx.x & 31;
  const int ty = threadIdx.x >> 5;          // 0..7
  const float* xp = x + ((long)b * CIN + c0) * NTOK + i0;
#pragma unroll
  for (int k = 0; k < 4; ++k)
    tile[ty + 8 * k][tx] = xp[(long)(ty + 8 * k) * NTOK + tx];
  __syncthreads();
  __bf16* op = Xt + ((long)b * NTOK + i0) * CIN + c0;
#pragma unroll
  for (int k = 0; k < 4; ++k)
    op[(long)(ty + 8 * k) * CIN + tx] = (__bf16)tile[tx][ty + 8 * k];
}

// ---------------------------------------------------------------------------
// Kernel 0b: one-time f32 -> bf16 conversion of both weight matrices.
// ---------------------------------------------------------------------------
__global__ void __launch_bounds__(256)
wconv_kernel(const float* __restrict__ Wf, const float* __restrict__ Wl,
             __bf16* __restrict__ Wfb, __bf16* __restrict__ Wlb) {
  const long nf = (long)QKVR * CIN / 4;     // float4 chunks in W_fc
  long idx = (long)blockIdx.x * blockDim.x + threadIdx.x;
  const float4* src;
  __bf16* dst;
  long o;
  if (idx < nf) { src = (const float4*)Wf; dst = Wfb; o = idx; }
  else          { src = (const float4*)Wl; dst = Wlb; o = idx - nf; }
  float4 v = src[o];
  union { __bf16 h[4]; uint2 u; } p;
  p.h[0] = (__bf16)v.x; p.h[1] = (__bf16)v.y;
  p.h[2] = (__bf16)v.z; p.h[3] = (__bf16)v.w;
  *(uint2*)(dst + o * 4) = p.u;
}

// ---------------------------------------------------------------------------
// Kernel 1: qkv = Xt @ Wfc^T + b_fc  (M=8192, N=1536, K=512), all-bf16 loads.
// Writes Q (b,h,n,D)*0.125, K (b,h,n,D), V transposed (b,h,D,n).
// Each wave: 16 rows x 64 cols. 12288 waves.
// ---------------------------------------------------------------------------
__global__ void __launch_bounds__(256)
qkv_kernel(const __bf16* __restrict__ Xt, const __bf16* __restrict__ Wfc,
           const float* __restrict__ b_fc,
           __bf16* __restrict__ Q, __bf16* __restrict__ Kb,
           __bf16* __restrict__ Vt) {
  const int lane  = threadIdx.x & 31;
  const int gwave = blockIdx.x * 8 + (threadIdx.x >> 5);
  const int tm  = gwave & 511;   // 512 M tiles of 16
  const int tnc = gwave >> 9;    // 24 chunks of 64 cols
  const int mbase = tm << 4;
  const int arow = mbase + (lane & 15);

  v8f acc[4] = {};
  for (int kk = 0; kk < CIN; kk += 32) {
    v16bf a = load_a_bf16(Xt, CIN, arow, kk, lane);
    if (kk + 32 < CIN)  // warm next A k-slice (global_prefetch_b8)
      __builtin_prefetch(Xt + (long)arow * CIN + kk + 32, 0, 1);
#pragma unroll
    for (int s = 0; s < 4; ++s) {
      int r = (tnc << 6) + (s << 4) + (lane & 15);
      v16bf bw = load_b_bf16(Wfc, CIN, r, kk, lane);
      acc[s] = wmma_bf16(a, bw, acc[s]);
    }
  }

  const int b = mbase >> 10;     // tiles never straddle a batch
  const int grp = lane >> 4;
#pragma unroll
  for (int s = 0; s < 4; ++s) {
    int r = (tnc << 6) + (s << 4) + (lane & 15);
    int head = r / 192;
    int t = r % 192;                 // 0..63 Q, 64..127 K, 128..191 V
    float bias = b_fc[r];
    long bh = (long)b * HEADS + head;
#pragma unroll
    for (int v = 0; v < 8; ++v) {
      int ii = (mbase + (grp << 3) + v) & 1023;
      float val = acc[s][v] + bias;
      if (t < DH) {
        Q[(bh * NTOK + ii) * DH + t] = (__bf16)(val * 0.125f);  // fold 1/sqrt(D)
      } else if (t < 2 * DH) {
        Kb[(bh * NTOK + ii) * DH + (t - DH)] = (__bf16)val;
      } else {
        Vt[(bh * DH + (t - 2 * DH)) * NTOK + ii] = (__bf16)val;
      }
    }
  }
}

// ---------------------------------------------------------------------------
// Kernel 2: flash attention, one wave per (b,h, 16-query tile). 4096 waves.
// Computes S^T = K.Q^T so P^T feeds O^T = V.P^T with only a shfl_xor(16)
// half-wave swap between the two WMMAs; online softmax state is per-lane.
// ---------------------------------------------------------------------------
__global__ void __launch_bounds__(256)
attn_kernel(const __bf16* __restrict__ Q, const __bf16* __restrict__ Kb,
            const __bf16* __restrict__ Vt, __bf16* __restrict__ O) {
  const int lane  = threadIdx.x & 31;
  const int gwave = blockIdx.x * 8 + (threadIdx.x >> 5);
  const int qt = gwave & 63;     // 64 query tiles
  const int bh = gwave >> 6;     // 64 (b,h) pairs
  const int b = bh >> 3, h = bh & 7;
  const __bf16* Qp = Q  + (long)bh * NTOK * DH;
  const __bf16* Kp = Kb + (long)bh * NTOK * DH;
  const __bf16* Vp = Vt + (long)bh * DH * NTOK;

  const int col = lane & 15;
  const int iq = (qt << 4) + col;          // this lane's query index
  // Q^T B-fragments for the two d-steps (live across the whole j loop)
  const v16bf bq0 = load_b_bf16(Qp, DH, iq, 0, lane);
  const v16bf bq1 = load_b_bf16(Qp, DH, iq, 32, lane);

  float mrun = -3.0e38f, lrun = 0.f;
  v8f acc[4] = {};                          // O^T: 4 d-tiles x (16 x 16)
  const int jr = lane & 15;
  const bool g1 = lane >= 16;

  for (int jb = 0; jb < NTOK; jb += 32) {
    // S^T tiles: rows j, cols i (two 16-row tiles, K-dim d = 64 in 2 steps)
    v8f t0 = {}, t1 = {};
    v16bf ak;
    ak = load_a_bf16(Kp, DH, jb + jr, 0, lane);       t0 = wmma_bf16(ak, bq0, t0);
    ak = load_a_bf16(Kp, DH, jb + jr, 32, lane);      t0 = wmma_bf16(ak, bq1, t0);
    ak = load_a_bf16(Kp, DH, jb + 16 + jr, 0, lane);  t1 = wmma_bf16(ak, bq0, t1);
    ak = load_a_bf16(Kp, DH, jb + 16 + jr, 32, lane); t1 = wmma_bf16(ak, bq1, t1);

    // per-query max over the 32 new scores (8 regs here + partner half-wave)
    float mx = t0[0];
#pragma unroll
    for (int v = 0; v < 8; ++v) { mx = fmaxf(mx, t0[v]); mx = fmaxf(mx, t1[v]); }
    mx = fmaxf(mx, __shfl_xor(mx, 16, 32));
    float mnew = fmaxf(mrun, mx);
    float alpha = __expf(mrun - mnew);

    float ssum = 0.f;
#pragma unroll
    for (int v = 0; v < 8; ++v) {
      t0[v] = __expf(t0[v] - mnew);
      t1[v] = __expf(t1[v] - mnew);
      ssum += t0[v] + t1[v];
    }
    ssum += __shfl_xor(ssum, 16, 32);
    lrun = lrun * alpha + ssum;
    mrun = mnew;
#pragma unroll
    for (int dt = 0; dt < 4; ++dt)
#pragma unroll
      for (int v = 0; v < 8; ++v) acc[dt][v] *= alpha;

    // Assemble P^T B-fragment from the S^T C/D tiles: half-wave swap only.
    float x0[8], x1[8];
#pragma unroll
    for (int v = 0; v < 8; ++v) {
      x0[v] = __shfl_xor(t0[v], 16, 32);
      x1[v] = __shfl_xor(t1[v], 16, 32);
    }
    v16bf pb;
#pragma unroll
    for (int e = 0; e < 8; ++e) {
      pb[e]     = (__bf16)(g1 ? x1[e] : t0[e]);
      pb[e + 8] = (__bf16)(g1 ? t1[e] : x0[e]);
    }

    // O^T += V(d, j) . P^T(j, i)   (V^T stored row-major in d -> contiguous A)
#pragma unroll
    for (int dt = 0; dt < 4; ++dt) {
      v16bf av = load_a_bf16(Vp, NTOK, (dt << 4) + jr, jb, lane);
      acc[dt] = wmma_bf16(av, pb, acc[dt]);
    }
  }

  const float inv = 1.f / lrun;
  const long orow = ((long)b * NTOK + iq) * HD + (long)h * DH + ((lane >> 4) << 3);
#pragma unroll
  for (int dt = 0; dt < 4; ++dt) {
    union { __bf16 hh[8]; uint4 u; } pk;
#pragma unroll
    for (int v = 0; v < 8; ++v) pk.hh[v] = (__bf16)(acc[dt][v] * inv);
    *(uint4*)(O + orow + (dt << 4)) = pk.u;   // 8 consecutive d -> 16B store
  }
}

// ---------------------------------------------------------------------------
// Kernel 3: out = O @ Wl^T + b_last + x^T, transposed back to (b,cin,h,w).
// M=8192, N=512, K=512; 16x64 per wave; residual fused, float4 I/O. 4096 waves.
// ---------------------------------------------------------------------------
__global__ void __launch_bounds__(256)
proj_kernel(const __bf16* __restrict__ O, const __bf16* __restrict__ Wl,
            const float* __restrict__ b_last, const float* __restrict__ x,
            float* __restrict__ out) {
  const int lane  = threadIdx.x & 31;
  const int gwave = blockIdx.x * 8 + (threadIdx.x >> 5);
  const int tm = gwave & 511;
  const int nc = gwave >> 9;          // 8 chunks of 64 output channels
  const int mbase = tm << 4;
  const int arow = mbase + (lane & 15);

  v8f acc[4] = {};
  for (int kk = 0; kk < HD; kk += 32) {
    v16bf a = load_a_bf16(O, HD, arow, kk, lane);
    if (kk + 32 < HD)
      __builtin_prefetch(O + (long)arow * HD + kk + 32, 0, 1);
#pragma unroll
    for (int s = 0; s < 4; ++s) {
      int c = (nc << 6) + (s << 4) + (lane & 15);
      v16bf bw = load_b_bf16(Wl, HD, c, kk, lane);
      acc[s] = wmma_bf16(a, bw, acc[s]);
    }
  }

  const int m0 = mbase + ((lane >> 4) << 3);
  const int b = m0 >> 10;
  const int i0 = m0 & 1023;           // 8-aligned -> float4 legal
#pragma unroll
  for (int s = 0; s < 4; ++s) {
    int c = (nc << 6) + (s << 4) + (lane & 15);
    float bias = b_last[c];
    long idx0 = ((long)b * CIN + c) * NTOK + i0;
    float4 r0 = *(const float4*)(x + idx0);
    float4 r1 = *(const float4*)(x + idx0 + 4);
    float4 o0, o1;
    o0.x = acc[s][0] + bias + r0.x;  o0.y = acc[s][1] + bias + r0.y;
    o0.z = acc[s][2] + bias + r0.z;  o0.w = acc[s][3] + bias + r0.w;
    o1.x = acc[s][4] + bias + r1.x;  o1.y = acc[s][5] + bias + r1.y;
    o1.z = acc[s][6] + bias + r1.z;  o1.w = acc[s][7] + bias + r1.w;
    *(float4*)(out + idx0)     = o0;
    *(float4*)(out + idx0 + 4) = o1;
  }
}

// ---------------------------------------------------------------------------
extern "C" void kernel_launch(void* const* d_in, const int* in_sizes, int n_in,
                              void* d_out, int out_size, void* d_ws, size_t ws_size,
                              hipStream_t stream) {
  const float* x      = (const float*)d_in[0];
  const float* W_fc   = (const float*)d_in[1];
  const float* b_fc   = (const float*)d_in[2];
  const float* W_last = (const float*)d_in[3];
  const float* b_last = (const float*)d_in[4];

  char* ws = (char*)d_ws;
  const size_t MB = 1024u * 1024u;
  __bf16* Q   = (__bf16*)(ws);                    // (b,h,n,D)  8 MB (pre-scaled)
  __bf16* Kb  = (__bf16*)(ws + 8 * MB);           // (b,h,n,D)  8 MB
  __bf16* Vt  = (__bf16*)(ws + 16 * MB);          // (b,h,D,n)  8 MB
  __bf16* Xt  = (__bf16*)(ws + 24 * MB);          // (b,n,cin)  8 MB
  __bf16* O   = (__bf16*)(ws + 24 * MB);          // aliases Xt (disjoint lifetime)
  __bf16* Wfb = (__bf16*)(ws + 32 * MB);          // 1536x512 bf16, 1.5 MB
  __bf16* Wlb = (__bf16*)(ws + 32 * MB + (size_t)QKVR * CIN * 2);  // 512x512, 0.5 MB

  // k0a: transpose+convert x -> Xt (LDS 32x32 tiles, both sides coalesced)
  xt_kernel<<<dim3(32, 16, 8), 256, 0, stream>>>(x, Xt);
  // k0b: weights f32 -> bf16 ((1536*512 + 512*512)/4 float4 chunks)
  wconv_kernel<<<1024, 256, 0, stream>>>(W_fc, W_last, Wfb, Wlb);
  // k1: 12288 waves (512 M-tiles x 24 column chunks)
  qkv_kernel<<<1536, 256, 0, stream>>>(Xt, Wfb, b_fc, Q, Kb, Vt);
  // k2: 4096 waves (64 (b,h) x 64 query tiles)
  attn_kernel<<<512, 256, 0, stream>>>(Q, Kb, Vt, O);
  // k3: 4096 waves (512 M-tiles x 8 column chunks)
  proj_kernel<<<512, 256, 0, stream>>>(O, Wlb, b_last, x, (float*)d_out);
}
